// GAT_63333587747452
// MI455X (gfx1250) — matrix operand order repaired and checked by reading, not compiled, not run on hardware
//
#include <hip/hip_runtime.h>

typedef __attribute__((ext_vector_type(16))) _Float16 v16h;
typedef __attribute__((ext_vector_type(8)))  _Float16 v8h;
typedef __attribute__((ext_vector_type(8)))  float    v8f;
typedef __attribute__((ext_vector_type(4)))  unsigned int u32x4;
typedef __attribute__((ext_vector_type(8)))  int      i32x8;
typedef __attribute__((ext_vector_type(4)))  int      i32x4;

constexpr int B_ = 16, N_ = 512, F_ = 64, T_ = 12, H_ = 4, O_ = 32, HO_ = 128, C_ = 64;
constexpr int NP_ = N_ + 16;    // padded LDS row stride (halves): keeps 32B align, kills bank conflicts
constexpr int FP_ = F_ + 16;
constexpr int HOP_ = HO_ + 16;
constexpr float NEGV  = -9e15f;
constexpr float ALPHA = 0.2f;

// ---- workspace layout (bytes) ----
constexpr size_t OFF_WA1  = 0;
constexpr size_t OFF_WA2  = OFF_WA1 + 1024;
constexpr size_t OFF_WAO1 = OFF_WA2 + 1024;
constexpr size_t OFF_WAO2 = OFF_WAO1 + 512;
constexpr size_t SZ_SCORE = (size_t)H_ * B_ * T_ * N_ * 4;
constexpr size_t OFF_WH1  = 4096;
constexpr size_t OFF_WH2  = OFF_WH1 + SZ_SCORE;
constexpr size_t OFF_WHF  = OFF_WH2 + SZ_SCORE;                 // f16 Wh, TRANSPOSED [h,b,t][o][n]
constexpr size_t SZ_WHF   = (size_t)H_ * B_ * T_ * N_ * O_ * 2;
constexpr size_t OFF_X1   = OFF_WHF + SZ_WHF;                   // f16 x1 [b,t][n][h*O]
constexpr size_t SZ_X1    = (size_t)B_ * T_ * N_ * HO_ * 2;
constexpr size_t SZ_SCO   = (size_t)B_ * T_ * N_ * 4;
constexpr size_t OFF_WH1O = OFF_X1 + SZ_X1;
constexpr size_t OFF_WH2O = OFF_WH1O + SZ_SCO;
constexpr size_t OFF_WHOF = OFF_WH2O + SZ_SCO;                  // f16 Wh_out, TRANSPOSED [b,t][c][n]

// --------------------------------------------------------------------------
// Tensor Data Mover: DMA a [nrows x 512] f16 row-major tile from global to
// LDS, inserting 32B padding after every 1024B row (-> padded stride NP_).
// D# layout per CDNA5 ISA ch.8. Issued by one wave; EXEC is ignored by TDM.
// --------------------------------------------------------------------------
__device__ __forceinline__ void tdm_load_rows(unsigned lds_off, const _Float16* gsrc,
                                              unsigned nrows) {
  unsigned long long ga = (unsigned long long)(size_t)gsrc;
  u32x4 g0;
  g0[0] = 1u;                                          // count=1, user descriptor
  g0[1] = lds_off;                                     // lds_addr (bytes)
  g0[2] = (unsigned)(ga & 0xFFFFFFFFu);                // global_addr[31:0]
  g0[3] = (unsigned)((ga >> 32) & 0x1FFFFFFu)          // global_addr[56:32]
        | 0x80000000u;                                 // type=2 ("image")
  i32x8 g1;
  g1[0] = (int)((1u << 16)      // data_size = 2 bytes
              | (1u << 20)      // pad_enable
              | (7u << 22)      // pad_interval: 256 DWORDs (=1024B row)
              | (7u << 25));    // pad_amount: 8 DWORDs (=32B)
  g1[1] = (int)(512u << 16);    // tensor_dim0 = 512 (low 16 in bits 63:48)
  g1[2] = (int)(nrows << 16);   // tensor_dim0 hi=0; tensor_dim1 = nrows
  g1[3] = (int)(512u << 16);    // tensor_dim1 hi=0; tile_dim0 = 512
  g1[4] = (int)nrows;           // tile_dim1 = nrows; tile_dim2 = 0
  g1[5] = 512;                  // tensor_dim0_stride = 512 elements
  g1[6] = 0;
  g1[7] = 0;
  i32x4 z4 = {0, 0, 0, 0};      // groups 2/3 unused (2D tensor)
  i32x8 z8 = {0, 0, 0, 0, 0, 0, 0, 0};
  __builtin_amdgcn_tensor_load_to_lds(g0, g1, z4, z4, z8, 0);   // 6-arg toolchain form
}

// --------------------------------------------------------------------------
// Kernel 0: fold attention vectors through W: wa = W @ a.
// --------------------------------------------------------------------------
__global__ void k_prep(const float* __restrict__ Whd, const float* __restrict__ ah,
                       const float* __restrict__ Wo,  const float* __restrict__ ao,
                       float* __restrict__ wa1, float* __restrict__ wa2,
                       float* __restrict__ wao1, float* __restrict__ wao2) {
  int tid = threadIdx.x;
  if (tid < H_ * F_) {
    int h = tid >> 6, f = tid & 63;
    float s1 = 0.f, s2 = 0.f;
    for (int o = 0; o < O_; ++o) {
      float w = Whd[((size_t)h * F_ + f) * O_ + o];
      s1 += w * ah[h * 2 * O_ + o];
      s2 += w * ah[h * 2 * O_ + O_ + o];
    }
    wa1[tid] = s1; wa2[tid] = s2;
  } else if (tid < H_ * F_ + HO_) {
    int k = tid - H_ * F_;
    float s1 = 0.f, s2 = 0.f;
    for (int c = 0; c < C_; ++c) {
      float w = Wo[(size_t)k * C_ + c];
      s1 += w * ao[c];
      s2 += w * ao[C_ + c];
    }
    wao1[k] = s1; wao2[k] = s2;
  }
}

// --------------------------------------------------------------------------
// Kernel 1: Wh = x @ W per (head,b,t) via WMMA; scores via folded vectors.
// Stores Wh TRANSPOSED [o][n] with packed 16B stores.
// grid = H*B*T*4, block = 256
// --------------------------------------------------------------------------
__global__ void __launch_bounds__(256)
k_wh(const float* __restrict__ x, const float* __restrict__ Whd,
     const float* __restrict__ wa1, const float* __restrict__ wa2,
     _Float16* __restrict__ whfT, float* __restrict__ wh1, float* __restrict__ wh2) {
  __shared__ _Float16 wT[O_ * FP_];   // W slice, transposed [c][f], padded
  int bid = blockIdx.x;
  int g  = bid & 3;
  int t  = (bid >> 2) % T_;
  int bt = (bid >> 2) / T_;
  int b  = bt % B_;
  int h  = bt / B_;

  for (int idx = threadIdx.x; idx < F_ * O_; idx += 256) {
    int f = idx >> 5, c = idx & 31;
    wT[c * FP_ + f] = (_Float16)Whd[(size_t)h * F_ * O_ + idx];
  }
  __syncthreads();

  int lane = threadIdx.x & 31;
  int wave = threadIdx.x >> 5;
  int n0 = (g * 8 + wave) * 16;
  int r  = lane & 15;
  int kbA = (lane < 16) ? 0 : 16;
  int kbB = (lane < 16) ? 0 : 16;
  int hi8 = (lane < 16) ? 0 : 8;
  int n = n0 + r;

  const float* xp = x + ((size_t)(b * N_ + n) * F_) * T_ + t;
  v16h af[2];
  float p1 = 0.f, p2 = 0.f;
#pragma unroll
  for (int kc = 0; kc < 2; ++kc) {
#pragma unroll
    for (int j = 0; j < 16; ++j) {
      int K = kc * 32 + kbA + j;
      float xv = xp[(size_t)K * T_];
      af[kc][j] = (_Float16)xv;
      p1 += xv * wa1[h * F_ + K];
      p2 += xv * wa2[h * F_ + K];
    }
  }
  p1 += __shfl_xor(p1, 16, 32);
  p2 += __shfl_xor(p2, 16, 32);
  size_t sl = ((size_t)(h * B_ + b) * T_ + t) * N_;
  if (lane < 16) { wh1[sl + n0 + lane] = p1; wh2[sl + n0 + lane] = p2; }

  size_t whbase = sl * O_;   // slice base in transposed [o][n] layout
#pragma unroll
  for (int oc = 0; oc < 2; ++oc) {
    int c = oc * 16 + (lane & 15);
    v8f acc = {};
#pragma unroll
    for (int kc = 0; kc < 2; ++kc) {
      v16h bf = *(const v16h*)(&wT[c * FP_ + kc * 32 + kbB]);
      acc = __builtin_amdgcn_wmma_f32_16x16x32_f16(false, af[kc], false, bf,
                                                   (short)0, acc, false, false);
    }
    v8h hv;
#pragma unroll
    for (int v = 0; v < 8; ++v) hv[v] = (_Float16)acc[v];
    *(v8h*)(&whfT[whbase + (size_t)c * N_ + n0 + hi8]) = hv;   // 16B packed store
  }
}

// --------------------------------------------------------------------------
// Kernel 2: fused masked softmax + att@Wh, ELU, head-major concat into x1.
// Wh tile (already transposed in global) DMA'd into padded LDS via TDM.
// grid = H*B*T*4, block = 256
// --------------------------------------------------------------------------
__global__ void __launch_bounds__(256)
k_attn1(const int* __restrict__ adj, const _Float16* __restrict__ whfT,
        const float* __restrict__ wh1g, const float* __restrict__ wh2g,
        _Float16* __restrict__ x1) {
  __shared__ _Float16 whT[O_ * NP_];   // ~33KB, [o][n] padded
  __shared__ float    wh2s[N_];
  int bid = blockIdx.x;
  int g  = bid & 3;
  int t  = (bid >> 2) % T_;
  int bt = (bid >> 2) / T_;
  int b  = bt % B_;
  int h  = bt / B_;
  size_t sl = ((size_t)(h * B_ + b) * T_ + t) * N_;

  if (threadIdx.x < 32) {
    tdm_load_rows((unsigned)(size_t)&whT[0], whfT + sl * O_, O_);
    __builtin_amdgcn_s_wait_tensorcnt(0);
  }
  for (int idx = threadIdx.x; idx < N_; idx += 256) wh2s[idx] = wh2g[sl + idx];
  __syncthreads();

  int lane = threadIdx.x & 31;
  int wave = threadIdx.x >> 5;
  int n0 = (g * 8 + wave) * 16;
  int r  = lane & 15;
  int n  = n0 + r;
  const int* arow = adj + (size_t)n * N_;
  float wh1r = wh1g[sl + n];
  int half = lane >> 4;

  float mx = -INFINITY;
  for (int i = 0; i < N_ / 2; ++i) {
    int m = 2 * i + half;
    float z = wh1r + wh2s[m];
    float e = (z > 0.f) ? z : ALPHA * z;
    e = (arow[m] > 0) ? e : NEGV;
    mx = fmaxf(mx, e);
  }
  mx = fmaxf(mx, __shfl_xor(mx, 16, 32));
  float sm = 0.f;
  for (int i = 0; i < N_ / 2; ++i) {
    int m = 2 * i + half;
    float z = wh1r + wh2s[m];
    float e = (z > 0.f) ? z : ALPHA * z;
    e = (arow[m] > 0) ? e : NEGV;
    sm += __expf(e - mx);
  }
  sm += __shfl_xor(sm, 16, 32);
  float inv = 1.f / sm;

  int kbA = (lane < 16) ? 0 : 16;
  int kbB = (lane < 16) ? 0 : 16;
  v8f acc[2] = {{}, {}};
  for (int k0 = 0; k0 < N_; k0 += 32) {
    v16h af;
#pragma unroll
    for (int j = 0; j < 16; ++j) {
      int m = k0 + kbA + j;
      float z = wh1r + wh2s[m];
      float e = (z > 0.f) ? z : ALPHA * z;
      e = (arow[m] > 0) ? e : NEGV;
      af[j] = (_Float16)(__expf(e - mx) * inv);
    }
#pragma unroll
    for (int oc = 0; oc < 2; ++oc) {
      int c = oc * 16 + (lane & 15);
      v16h bf = *(const v16h*)(&whT[c * NP_ + k0 + kbB]);
      acc[oc] = __builtin_amdgcn_wmma_f32_16x16x32_f16(false, af, false, bf,
                                                       (short)0, acc[oc], false, false);
    }
  }
  size_t x1base = ((size_t)(b * T_ + t) * N_) * HO_;
#pragma unroll
  for (int oc = 0; oc < 2; ++oc) {
    int c = oc * 16 + (lane & 15);
#pragma unroll
    for (int v = 0; v < 8; ++v) {
      int row = v + ((lane < 16) ? 0 : 8);
      float hv = acc[oc][v];
      float ev = (hv > 0.f) ? hv : (__expf(hv) - 1.f);
      x1[x1base + (size_t)(n0 + row) * HO_ + h * O_ + c] = (_Float16)ev;
    }
  }
}

// --------------------------------------------------------------------------
// Kernel 3: Wh_out = x1 @ W_out via WMMA + layer-2 scores; stores transposed.
// grid = B*T*4, block = 256
// --------------------------------------------------------------------------
__global__ void __launch_bounds__(256)
k_whout(const _Float16* __restrict__ x1, const float* __restrict__ Wo,
        const float* __restrict__ wao1, const float* __restrict__ wao2,
        _Float16* __restrict__ whofT, float* __restrict__ wh1o, float* __restrict__ wh2o) {
  __shared__ _Float16 woT[C_ * HOP_];  // 18KB, transposed [c][k], padded
  int bid = blockIdx.x;
  int g = bid & 3;
  int t = (bid >> 2) % T_;
  int b = (bid >> 2) / T_;
  for (int idx = threadIdx.x; idx < HO_ * C_; idx += 256) {
    int k = idx >> 6, c = idx & 63;
    woT[c * HOP_ + k] = (_Float16)Wo[idx];
  }
  __syncthreads();

  int lane = threadIdx.x & 31;
  int wave = threadIdx.x >> 5;
  int n0 = (g * 8 + wave) * 16;
  int r  = lane & 15;
  int n  = n0 + r;
  size_t sl = ((size_t)b * T_ + t) * N_;
  const _Float16* xp = x1 + (sl + n) * HO_;

  int kbA = (lane < 16) ? 0 : 16;
  int kbB = (lane < 16) ? 0 : 16;
  int hi8 = (lane < 16) ? 0 : 8;
  v16h af[4];
  float p1 = 0.f, p2 = 0.f;
#pragma unroll
  for (int kc = 0; kc < 4; ++kc) {
    v8h lo = *(const v8h*)(xp + kc * 32 + kbA);        // 16B vector loads
    v8h hg = *(const v8h*)(xp + kc * 32 + kbA + 8);
#pragma unroll
    for (int j = 0; j < 8; ++j) {
      af[kc][j]     = lo[j];
      af[kc][j + 8] = hg[j];
      int K = kc * 32 + kbA + j;
      p1 += (float)lo[j] * wao1[K]      + (float)hg[j] * wao1[K + 8];
      p2 += (float)lo[j] * wao2[K]      + (float)hg[j] * wao2[K + 8];
    }
  }
  p1 += __shfl_xor(p1, 16, 32);
  p2 += __shfl_xor(p2, 16, 32);
  if (lane < 16) { wh1o[sl + n0 + lane] = p1; wh2o[sl + n0 + lane] = p2; }

  size_t whbase = sl * C_;   // transposed [c][n] slice base
#pragma unroll
  for (int oc = 0; oc < 4; ++oc) {
    int c = oc * 16 + (lane & 15);
    v8f acc = {};
#pragma unroll
    for (int kc = 0; kc < 4; ++kc) {
      v16h bf = *(const v16h*)(&woT[c * HOP_ + kc * 32 + kbB]);
      acc = __builtin_amdgcn_wmma_f32_16x16x32_f16(false, af[kc], false, bf,
                                                   (short)0, acc, false, false);
    }
    v8h hv;
#pragma unroll
    for (int v = 0; v < 8; ++v) hv[v] = (_Float16)acc[v];
    *(v8h*)(&whofT[whbase + (size_t)c * N_ + n0 + hi8]) = hv;
  }
}

// --------------------------------------------------------------------------
// Kernel 4: layer-2 fused masked softmax + att@Wh_out; fp32 out [B,N,C,T].
// 64KB Wh_out tile DMA'd via TDM into padded LDS. grid = B*T*4, block = 256
// --------------------------------------------------------------------------
__global__ void __launch_bounds__(256)
k_attn2(const int* __restrict__ adj, const _Float16* __restrict__ whofT,
        const float* __restrict__ wh1o, const float* __restrict__ wh2o,
        float* __restrict__ out) {
  __shared__ _Float16 whT[C_ * NP_];   // ~66KB, [c][n] padded (WGP LDS: 320KB)
  __shared__ float    wh2s[N_];
  int bid = blockIdx.x;
  int g = bid & 3;
  int t = (bid >> 2) % T_;
  int b = (bid >> 2) / T_;
  size_t sl = ((size_t)b * T_ + t) * N_;

  if (threadIdx.x < 32) {
    tdm_load_rows((unsigned)(size_t)&whT[0], whofT + sl * C_, C_);
    __builtin_amdgcn_s_wait_tensorcnt(0);
  }
  for (int idx = threadIdx.x; idx < N_; idx += 256) wh2s[idx] = wh2o[sl + idx];
  __syncthreads();

  int lane = threadIdx.x & 31;
  int wave = threadIdx.x >> 5;
  int n0 = (g * 8 + wave) * 16;
  int r  = lane & 15;
  int n  = n0 + r;
  const int* arow = adj + (size_t)n * N_;
  float wh1r = wh1o[sl + n];
  int half = lane >> 4;

  float mx = -INFINITY;
  for (int i = 0; i < N_ / 2; ++i) {
    int m = 2 * i + half;
    float z = wh1r + wh2s[m];
    float e = (z > 0.f) ? z : ALPHA * z;
    e = (arow[m] > 0) ? e : NEGV;
    mx = fmaxf(mx, e);
  }
  mx = fmaxf(mx, __shfl_xor(mx, 16, 32));
  float sm = 0.f;
  for (int i = 0; i < N_ / 2; ++i) {
    int m = 2 * i + half;
    float z = wh1r + wh2s[m];
    float e = (z > 0.f) ? z : ALPHA * z;
    e = (arow[m] > 0) ? e : NEGV;
    sm += __expf(e - mx);
  }
  sm += __shfl_xor(sm, 16, 32);
  float inv = 1.f / sm;

  int kbA = (lane < 16) ? 0 : 16;
  int kbB = (lane < 16) ? 0 : 16;
  v8f acc[4] = {{}, {}, {}, {}};
  for (int k0 = 0; k0 < N_; k0 += 32) {
    v16h af;
#pragma unroll
    for (int j = 0; j < 16; ++j) {
      int m = k0 + kbA + j;
      float z = wh1r + wh2s[m];
      float e = (z > 0.f) ? z : ALPHA * z;
      e = (arow[m] > 0) ? e : NEGV;
      af[j] = (_Float16)(__expf(e - mx) * inv);
    }
#pragma unroll
    for (int oc = 0; oc < 4; ++oc) {
      int c = oc * 16 + (lane & 15);
      v16h bf = *(const v16h*)(&whT[c * NP_ + k0 + kbB]);
      acc[oc] = __builtin_amdgcn_wmma_f32_16x16x32_f16(false, af, false, bf,
                                                       (short)0, acc[oc], false, false);
    }
  }
  // no ELU (concat=False); transpose to [B,N,C,T]
#pragma unroll
  for (int oc = 0; oc < 4; ++oc) {
    int c = oc * 16 + (lane & 15);
#pragma unroll
    for (int v = 0; v < 8; ++v) {
      int row = v + ((lane < 16) ? 0 : 8);
      out[(((size_t)b * N_ + n0 + row) * C_ + c) * T_ + t] = acc[oc][v];
    }
  }
}

// --------------------------------------------------------------------------
extern "C" void kernel_launch(void* const* d_in, const int* in_sizes, int n_in,
                              void* d_out, int out_size, void* d_ws, size_t ws_size,
                              hipStream_t stream) {
  (void)in_sizes; (void)n_in; (void)out_size; (void)ws_size;
  const float* x   = (const float*)d_in[0];
  const int*   adj = (const int*)d_in[1];
  const float* Whd = (const float*)d_in[2];
  const float* ah  = (const float*)d_in[3];
  const float* Wo  = (const float*)d_in[4];
  const float* ao  = (const float*)d_in[5];

  char* ws = (char*)d_ws;
  float*    wa1   = (float*)(ws + OFF_WA1);
  float*    wa2   = (float*)(ws + OFF_WA2);
  float*    wao1  = (float*)(ws + OFF_WAO1);
  float*    wao2  = (float*)(ws + OFF_WAO2);
  float*    wh1   = (float*)(ws + OFF_WH1);
  float*    wh2   = (float*)(ws + OFF_WH2);
  _Float16* whfT  = (_Float16*)(ws + OFF_WHF);
  _Float16* x1f   = (_Float16*)(ws + OFF_X1);
  float*    wh1o  = (float*)(ws + OFF_WH1O);
  float*    wh2o  = (float*)(ws + OFF_WH2O);
  _Float16* whofT = (_Float16*)(ws + OFF_WHOF);

  k_prep  <<<1, 384, 0, stream>>>(Whd, ah, Wo, ao, wa1, wa2, wao1, wao2);
  k_wh    <<<H_ * B_ * T_ * 4, 256, 0, stream>>>(x, Whd, wa1, wa2, whfT, wh1, wh2);
  k_attn1 <<<H_ * B_ * T_ * 4, 256, 0, stream>>>(adj, whfT, wh1, wh2, x1f);
  k_whout <<<B_ * T_ * 4, 256, 0, stream>>>(x1f, Wo, wao1, wao2, whofT, wh1o, wh2o);
  k_attn2 <<<B_ * T_ * 4, 256, 0, stream>>>(adj, whofT, wh1o, wh2o, (float*)d_out);
}
